// DeeperGCNLayer_14697378087223
// MI455X (gfx1250) — compile-verified, matrix operand order checked by reading, not run on hardware
//
#include <hip/hip_runtime.h>

typedef __attribute__((ext_vector_type(16))) _Float16 v16h;
typedef __attribute__((ext_vector_type(8)))  _Float16 v8h;
typedef __attribute__((ext_vector_type(4)))  _Float16 v4h;
typedef __attribute__((ext_vector_type(8)))  float    v8f;

#define NNODES 50000
#define FD     128      // D
#define FD2    256      // 2D
#define MSG_EPS 1e-7f
#define SM_EPS  1e-16f
#define BN_EPS  1e-5f
#define LN_EPS  1e-5f

// Load a WMMA 16-half fragment from LDS: two contiguous 8-half groups at
// p[0..7] and p[16..23] (both 16-byte aligned) -> 2x ds_load_b128.
static __device__ __forceinline__ v16h load_frag(const _Float16* p) {
    v8h lo = *(const v8h*)(p);
    v8h hi = *(const v8h*)(p + 16);
    return __builtin_shufflevector(lo, hi, 0, 1, 2, 3, 4, 5, 6, 7,
                                           8, 9, 10, 11, 12, 13, 14, 15);
}

static __device__ __forceinline__ v4h cvt4(float a, float b, float c, float d) {
    v4h r;
    r[0] = (_Float16)a; r[1] = (_Float16)b;
    r[2] = (_Float16)c; r[3] = (_Float16)d;
    return r;
}

// ---------------------------------------------------------------- zero fill
__global__ void zero_kernel(float4* __restrict__ p, long n4) {
    long i = (long)blockIdx.x * blockDim.x + threadIdx.x;
    long stride = (long)gridDim.x * blockDim.x;
    float4 z; z.x = 0.f; z.y = 0.f; z.z = 0.f; z.w = 0.f;
    for (; i < n4; i += stride) p[i] = z;
}

// --------------------------------------------- edge pass 1: segment max (L2 atomics)
// one wave per edge; 32 lanes x float4 = 128 features, coalesced row load.
__global__ __launch_bounds__(256)
void edge_max_kernel(const float* __restrict__ x, const int* __restrict__ src,
                     const int* __restrict__ dst, const float* __restrict__ tptr,
                     float* __restrict__ maxb, int E)
{
    int wid  = (int)(((long)blockIdx.x * blockDim.x + threadIdx.x) >> 5);
    if (wid >= E) return;
    int lane = threadIdx.x & 31;
    float t  = tptr[0];
    int s = src[wid];
    int d = dst[wid];
    const float4 v = ((const float4*)(x + (size_t)s * FD))[lane];
    float l0 = t * (fmaxf(v.x, 0.f) + MSG_EPS);
    float l1 = t * (fmaxf(v.y, 0.f) + MSG_EPS);
    float l2 = t * (fmaxf(v.z, 0.f) + MSG_EPS);
    float l3 = t * (fmaxf(v.w, 0.f) + MSG_EPS);
    int* mb = (int*)(maxb + (size_t)d * FD) + lane * 4;
    atomicMax(mb + 0, __float_as_int(l0));
    atomicMax(mb + 1, __float_as_int(l1));
    atomicMax(mb + 2, __float_as_int(l2));
    atomicMax(mb + 3, __float_as_int(l3));
}

// --------------------------------------------- edge pass 2: exp sums (fused num+den)
__global__ __launch_bounds__(256)
void edge_sum_kernel(const float* __restrict__ x, const int* __restrict__ src,
                     const int* __restrict__ dst, const float* __restrict__ tptr,
                     const float* __restrict__ maxb,
                     float* __restrict__ numb, float* __restrict__ denb, int E)
{
    int wid  = (int)(((long)blockIdx.x * blockDim.x + threadIdx.x) >> 5);
    if (wid >= E) return;
    int lane = threadIdx.x & 31;
    float t  = tptr[0];
    int s = src[wid];
    int d = dst[wid];
    const float4 v  = ((const float4*)(x    + (size_t)s * FD))[lane];
    const float4 mx = ((const float4*)(maxb + (size_t)d * FD))[lane];
    float m0 = fmaxf(v.x, 0.f) + MSG_EPS;
    float m1 = fmaxf(v.y, 0.f) + MSG_EPS;
    float m2 = fmaxf(v.z, 0.f) + MSG_EPS;
    float m3 = fmaxf(v.w, 0.f) + MSG_EPS;
    float e0 = __expf(t * m0 - mx.x);
    float e1 = __expf(t * m1 - mx.y);
    float e2 = __expf(t * m2 - mx.z);
    float e3 = __expf(t * m3 - mx.w);
    float* db = denb + (size_t)d * FD + lane * 4;
    float* nb = numb + (size_t)d * FD + lane * 4;
    atomicAdd(db + 0, e0);       atomicAdd(db + 1, e1);
    atomicAdd(db + 2, e2);       atomicAdd(db + 3, e3);
    atomicAdd(nb + 0, e0 * m0);  atomicAdd(nb + 1, e1 * m1);
    atomicAdd(nb + 2, e2 * m2);  atomicAdd(nb + 3, e3 * m3);
}

// --------------------------------------------- GEMM1: h1 = (x + agg) @ W1 + b1, + BN stats
// WG = 128 threads (4 waves), computes 16 rows x 256 cols.
// W1 staged TRANSPOSED in LDS f16 (Wt[n][k]) so B fragments vectorize to b128.
__global__ __launch_bounds__(128)
void gemm1_kernel(const float* __restrict__ x,
                  const float* __restrict__ numb, const float* __restrict__ denb,
                  const float* __restrict__ W1,  const float* __restrict__ b1,
                  float* __restrict__ h1,
                  float* __restrict__ colsum, float* __restrict__ colsumsq)
{
    __shared__ _Float16 Wt[FD2][64 + 8];  // 36 KB: one K-half of W1, transposed
    __shared__ _Float16 As[16][FD + 8];   // h0 tile, pitch 136 halves (272 B)

    const int row0 = blockIdx.x * 16;
    const int tid  = threadIdx.x;
    const int lane = tid & 31;
    const int wave = tid >> 5;

    // stage A: h0 = x + num/(den+eps), 16x128 f16, float4 granularity (512 vec4)
    for (int i = tid; i < 16 * (FD / 4); i += 128) {
        int m = i >> 5, c4 = i & 31;            // 32 float4 per row
        size_t idx4 = ((size_t)(row0 + m) * FD) / 4 + c4;
        float4 xv = ((const float4*)x)[idx4];
        float4 nv = ((const float4*)numb)[idx4];
        float4 dv = ((const float4*)denb)[idx4];
        *(v4h*)&As[m][c4 * 4] = cvt4(xv.x + nv.x / (dv.x + SM_EPS),
                                     xv.y + nv.y / (dv.y + SM_EPS),
                                     xv.z + nv.z / (dv.z + SM_EPS),
                                     xv.w + nv.w / (dv.w + SM_EPS));
    }

    v8f acc[4];
    #pragma unroll
    for (int i = 0; i < 4; ++i) acc[i] = (v8f){0.f,0.f,0.f,0.f,0.f,0.f,0.f,0.f};

    const int mlane = lane & 15;
    const int koff  = (lane >> 4) * 8;

    for (int kh = 0; kh < 2; ++kh) {
        __syncthreads();
        // stage W1 rows [kh*64, kh*64+64) transposed: float4 global, b16 LDS scatter
        for (int i = tid; i < 64 * (FD2 / 4); i += 128) {
            int k = i >> 6, n4 = i & 63;        // 64 float4 per W1 row
            float4 w = ((const float4*)W1)[((size_t)(kh * 64 + k) * FD2) / 4 + n4];
            int n = n4 * 4;
            Wt[n + 0][k] = (_Float16)w.x;
            Wt[n + 1][k] = (_Float16)w.y;
            Wt[n + 2][k] = (_Float16)w.z;
            Wt[n + 3][k] = (_Float16)w.w;
        }
        __syncthreads();

        for (int kc = 0; kc < 64; kc += 32) {
            v16h a = load_frag(&As[mlane][kh * 64 + kc + koff]);
            #pragma unroll
            for (int tcol = 0; tcol < 4; ++tcol) {
                int n = wave * 64 + tcol * 16 + mlane;
                v16h b = load_frag(&Wt[n][kc + koff]);
                acc[tcol] = __builtin_amdgcn_wmma_f32_16x16x32_f16(
                    false, a, false, b, (short)0, acc[tcol], false, false);
            }
        }
    }

    // epilogue: +b1, store h1, per-column partial sums for BN
    const int mbase = (lane >> 4) * 8;
    #pragma unroll
    for (int tcol = 0; tcol < 4; ++tcol) {
        int n = wave * 64 + tcol * 16 + mlane;
        float bias = b1[n];
        float s = 0.f, sq = 0.f;
        #pragma unroll
        for (int r = 0; r < 8; ++r) {
            float v = acc[tcol][r] + bias;
            h1[(size_t)(row0 + mbase + r) * FD2 + n] = v;
            s += v; sq += v * v;
        }
        atomicAdd(&colsum[n], s);
        atomicAdd(&colsumsq[n], sq);
    }
}

// --------------------------------------------- BN finalize: fold mu/var/gamma/beta
__global__ void bnstats_kernel(const float* __restrict__ colsum,
                               const float* __restrict__ colsumsq,
                               const float* __restrict__ gamma,
                               const float* __restrict__ beta,
                               float* __restrict__ bnA, float* __restrict__ bnC)
{
    int n = threadIdx.x;                 // 256 threads, 1 block
    float inv_n = 1.0f / (float)NNODES;
    float mu  = colsum[n] * inv_n;
    float var = colsumsq[n] * inv_n - mu * mu;
    float a   = gamma[n] * rsqrtf(var + BN_EPS);
    bnA[n] = a;
    bnC[n] = beta[n] - mu * a;
}

// --------------------------------------------- GEMM2 + LN + ReLU + residual
// WG = 128 threads, 16 rows x 128 cols. A = relu(BN(h1)) staged f16; W2 transposed in LDS.
__global__ __launch_bounds__(128)
void gemm2_kernel(const float* __restrict__ x,  const float* __restrict__ h1,
                  const float* __restrict__ bnA, const float* __restrict__ bnC,
                  const float* __restrict__ W2,  const float* __restrict__ b2,
                  const float* __restrict__ lng, const float* __restrict__ lnb,
                  float* __restrict__ out)
{
    __shared__ _Float16 Wt[FD][128 + 8];   // 34 KB: one K-half of W2, transposed
    __shared__ _Float16 As[16][FD2 + 8];   // A tile, pitch 264 halves (528 B)
    __shared__ float    Rs[16][FD];        // result tile
    __shared__ float    red[16][8][2];
    __shared__ float    stat[16][2];

    const int row0 = blockIdx.x * 16;
    const int tid  = threadIdx.x;
    const int lane = tid & 31;
    const int wave = tid >> 5;

    // stage A: relu(h1 * a + c), 16x256 f16, float4 granularity (1024 vec4)
    for (int i = tid; i < 16 * (FD2 / 4); i += 128) {
        int m = i >> 6, c4 = i & 63;            // 64 float4 per row
        float4 hv = ((const float4*)h1)[((size_t)(row0 + m) * FD2) / 4 + c4];
        float4 av = ((const float4*)bnA)[c4];
        float4 cv = ((const float4*)bnC)[c4];
        *(v4h*)&As[m][c4 * 4] = cvt4(fmaxf(hv.x * av.x + cv.x, 0.f),
                                     fmaxf(hv.y * av.y + cv.y, 0.f),
                                     fmaxf(hv.z * av.z + cv.z, 0.f),
                                     fmaxf(hv.w * av.w + cv.w, 0.f));
    }

    v8f acc[2];
    acc[0] = (v8f){0.f,0.f,0.f,0.f,0.f,0.f,0.f,0.f};
    acc[1] = (v8f){0.f,0.f,0.f,0.f,0.f,0.f,0.f,0.f};

    const int mlane = lane & 15;
    const int koff  = (lane >> 4) * 8;

    for (int kh = 0; kh < 2; ++kh) {
        __syncthreads();
        // stage W2 rows [kh*128, kh*128+128) transposed
        for (int i = tid; i < 128 * (FD / 4); i += 128) {
            int k = i >> 5, n4 = i & 31;        // 32 float4 per W2 row
            float4 w = ((const float4*)W2)[((size_t)(kh * 128 + k) * FD) / 4 + n4];
            int n = n4 * 4;
            Wt[n + 0][k] = (_Float16)w.x;
            Wt[n + 1][k] = (_Float16)w.y;
            Wt[n + 2][k] = (_Float16)w.z;
            Wt[n + 3][k] = (_Float16)w.w;
        }
        __syncthreads();

        for (int kc = 0; kc < 128; kc += 32) {
            v16h a = load_frag(&As[mlane][kh * 128 + kc + koff]);
            #pragma unroll
            for (int tcol = 0; tcol < 2; ++tcol) {
                int n = wave * 32 + tcol * 16 + mlane;
                v16h b = load_frag(&Wt[n][kc + koff]);
                acc[tcol] = __builtin_amdgcn_wmma_f32_16x16x32_f16(
                    false, a, false, b, (short)0, acc[tcol], false, false);
            }
        }
    }

    // write tile (+b2) to LDS
    const int mbase = (lane >> 4) * 8;
    #pragma unroll
    for (int tcol = 0; tcol < 2; ++tcol) {
        int n = wave * 32 + tcol * 16 + mlane;
        float bias = b2[n];
        #pragma unroll
        for (int r = 0; r < 8; ++r)
            Rs[mbase + r][n] = acc[tcol][r] + bias;
    }
    __syncthreads();

    // LayerNorm per row (128 cols): 8 partial sums per row
    {
        int row = tid & 15, j = tid >> 4;
        float s = 0.f, sq = 0.f;
        #pragma unroll
        for (int c = 0; c < 16; ++c) {
            float v = Rs[row][j * 16 + c];
            s += v; sq += v * v;
        }
        red[row][j][0] = s; red[row][j][1] = sq;
    }
    __syncthreads();
    if (tid < 16) {
        float s = 0.f, sq = 0.f;
        #pragma unroll
        for (int j = 0; j < 8; ++j) { s += red[tid][j][0]; sq += red[tid][j][1]; }
        float mu  = s * (1.0f / FD);
        float var = sq * (1.0f / FD) - mu * mu;
        stat[tid][0] = mu;
        stat[tid][1] = rsqrtf(var + LN_EPS);
    }
    __syncthreads();

    // apply LN, ReLU, residual; write out (vectorized: float4 lanes)
    {
        int m = tid >> 3, cb = (tid & 7) * 16;
        float mu = stat[m][0], inv = stat[m][1];
        size_t base4 = ((size_t)(row0 + m) * FD + cb) / 4;
        #pragma unroll
        for (int q = 0; q < 4; ++q) {
            int c = cb + q * 4;
            float4 g = ((const float4*)lng)[c / 4];
            float4 bt = ((const float4*)lnb)[c / 4];
            float4 xv = ((const float4*)x)[base4 + q];
            float4 o;
            o.x = xv.x + fmaxf((Rs[m][c + 0] - mu) * inv * g.x + bt.x, 0.f);
            o.y = xv.y + fmaxf((Rs[m][c + 1] - mu) * inv * g.y + bt.y, 0.f);
            o.z = xv.z + fmaxf((Rs[m][c + 2] - mu) * inv * g.z + bt.z, 0.f);
            o.w = xv.w + fmaxf((Rs[m][c + 3] - mu) * inv * g.w + bt.w, 0.f);
            ((float4*)out)[base4 + q] = o;
        }
    }
}

// ---------------------------------------------------------------- launch
extern "C" void kernel_launch(void* const* d_in, const int* in_sizes, int n_in,
                              void* d_out, int out_size, void* d_ws, size_t ws_size,
                              hipStream_t stream)
{
    const float* x    = (const float*)d_in[0];
    const int*   ei   = (const int*)  d_in[1];
    const float* tptr = (const float*)d_in[2];
    const float* W1   = (const float*)d_in[3];
    const float* b1   = (const float*)d_in[4];
    const float* bng  = (const float*)d_in[5];
    const float* bnb  = (const float*)d_in[6];
    const float* W2   = (const float*)d_in[7];
    const float* b2   = (const float*)d_in[8];
    const float* lng  = (const float*)d_in[9];
    const float* lnb  = (const float*)d_in[10];
    float* out = (float*)d_out;
    float* ws  = (float*)d_ws;

    const int E = in_sizes[1] / 2;
    const int* src = ei;
    const int* dst = ei + E;

    const size_t nd = (size_t)NNODES * FD;      // 6.4M floats
    // Atomic accumulators first (zeroed each call), then non-zeroed scratch.
    float* maxb = ws;
    float* numb = maxb + nd;
    float* denb = numb + nd;
    float* csum = denb + nd;
    float* csq  = csum + FD2;
    float* bnA  = csq  + FD2;
    float* bnC  = bnA  + FD2;
    float* h1   = bnC  + FD2;                   // N x 256 (fully overwritten)

    // 1) zero atomic accumulators only: 3*nd + 1024 floats (multiple of 4)
    long zero_floats = (long)(3 * nd) + 4 * FD2;
    zero_kernel<<<2048, 256, 0, stream>>>((float4*)ws, zero_floats / 4);

    // 2) edge passes (one wave / edge)
    int eblocks = (E + 7) / 8;
    edge_max_kernel<<<eblocks, 256, 0, stream>>>(x, src, dst, tptr, maxb, E);
    edge_sum_kernel<<<eblocks, 256, 0, stream>>>(x, src, dst, tptr, maxb, numb, denb, E);

    // 3) GEMM1 + BN stats  (50000 / 16 = 3125 tiles exactly)
    gemm1_kernel<<<NNODES / 16, 128, 0, stream>>>(x, numb, denb, W1, b1, h1, csum, csq);

    // 4) BN finalize
    bnstats_kernel<<<1, FD2, 0, stream>>>(csum, csq, bng, bnb, bnA, bnC);

    // 5) GEMM2 + LN + ReLU + residual
    gemm2_kernel<<<NNODES / 16, 128, 0, stream>>>(x, h1, bnA, bnC, W2, b2, lng, lnb, out);
}
